// Attention_23708219474328
// MI455X (gfx1250) — compile-verified, matrix operand order checked by reading, not run on hardware
//
#include <hip/hip_runtime.h>
#include <hip/hip_bf16.h>

// ---- problem constants (match reference) ----
#define DMODEL   2048
#define KVDIM    512        // N_KV_HEADS * HEAD_DIM
#define HEAD_DIM 64
#define N_HEADS  32
#define SEQ      2048
#define BATCH    2
#define BS       (BATCH * SEQ)   // 4096 rows

typedef __attribute__((ext_vector_type(16))) __bf16 v16bf;
typedef __attribute__((ext_vector_type(2)))  __bf16 v2bf;
typedef __attribute__((ext_vector_type(8)))  float  v8f;

// native bf16 converts (backend selects gfx1250 hw cvt, fuses into packed form)
__device__ __forceinline__ unsigned short bfu(float a) {
  union { __bf16 b; unsigned short u; } o; o.b = (__bf16)a; return o.u;
}
__device__ __forceinline__ unsigned pk_bf16(float a, float b) {
  union { v2bf v; unsigned u; } o;
  o.v[0] = (__bf16)a; o.v[1] = (__bf16)b;
  return o.u;
}

__device__ __forceinline__ v8f wmma_bf16(v16bf a, v16bf b, v8f c) {
  // D = A(16x32 bf16) * B(32x16 bf16) + C(16x16 f32)
  return __builtin_amdgcn_wmma_f32_16x16x32_bf16(false, a, false, b, (short)0, c, false, false);
}

// 16x32 bf16 A-style fragment load (B operands are transposed matrices here,
// so they use the identical row-contiguous pattern).
// ISA layout: lane<16 -> row lane&15, K = kbase+[0..7] then kbase+16+[0..7];
// lane>=16 -> same rows, K shifted +8.
__device__ __forceinline__ v16bf load_frag(const unsigned short* base, int row0,
                                           int stride, int kbase, int lane) {
  int r  = row0 + (lane & 15);
  int k0 = kbase + ((lane >> 4) << 3);
  const unsigned short* p = base + (size_t)r * stride + k0;
  union { uint4 u[2]; v16bf v; } f;
  f.u[0] = *(const uint4*)(p);
  f.u[1] = *(const uint4*)(p + 16);
  return f.v;
}

// ---------------------------------------------------------------------------
// C[M,N] = A[M,K] @ W[N,K]^T   (bf16 WMMA, f32 accumulate)
// Block: 256 threads (8 waves, 2x4), tile 128(M) x 256(N), BK=64.
// Each wave owns a 64x64 C tile: 16 v8f accumulators, 16 WMMAs per k-step
// fed by only 8 fragment loads (4 A + 4 B, each reused 4x).
// ---------------------------------------------------------------------------
template<bool A_IS_F32, bool OUT_IS_F32>
__global__ __launch_bounds__(256) void gemm_wmma_kernel(
    const void* __restrict__ Aptr, const float* __restrict__ W,
    void* __restrict__ Cptr, int M, int N, int K)
{
  __shared__ unsigned short sA[128 * 72];   // +8 pad breaks bank conflicts
  __shared__ unsigned short sB[256 * 72];

  const int t = threadIdx.x;
  const int lane = t & 31;
  const int wave = t >> 5;
  const int waveM = wave >> 2;   // 0..1 -> 64-row slab
  const int waveN = wave & 3;    // 0..3 -> 64-col slab
  const int m0 = blockIdx.y * 128;
  const int n0 = blockIdx.x * 256;

  const v8f vzero = {};
  v8f acc[4][4];
#pragma unroll
  for (int i = 0; i < 4; ++i)
#pragma unroll
    for (int j = 0; j < 4; ++j) acc[i][j] = vzero;

  for (int kt = 0; kt < K; kt += 64) {
    __syncthreads();
    // ---- stage A tile (128x64) ----
#pragma unroll
    for (int i = 0; i < 8; ++i) {
      int idx = i * 1024 + t * 4;
      int row = idx >> 6, col = idx & 63;
      uint2 d;
      if constexpr (A_IS_F32) {
        const float* Af = (const float*)Aptr;
        const float* ap = Af + (size_t)(m0 + row) * K + kt + col;
        float4 f = *(const float4*)ap;
        d.x = pk_bf16(f.x, f.y);
        d.y = pk_bf16(f.z, f.w);
        if (kt + 64 < K) __builtin_prefetch(ap + 64, 0, 1);  // next K-tile
      } else {
        const unsigned short* Ab = (const unsigned short*)Aptr;
        const unsigned short* ap = Ab + (size_t)(m0 + row) * K + kt + col;
        d = *(const uint2*)ap;
        if (kt + 64 < K) __builtin_prefetch(ap + 64, 0, 1);
      }
      *(uint2*)(&sA[row * 72 + col]) = d;
    }
    // ---- stage W tile (256x64) ----
#pragma unroll
    for (int i = 0; i < 16; ++i) {
      int idx = i * 1024 + t * 4;
      int row = idx >> 6, col = idx & 63;
      const float* wp = W + (size_t)(n0 + row) * K + kt + col;
      float4 g = *(const float4*)wp;
      uint2 e;
      e.x = pk_bf16(g.x, g.y);
      e.y = pk_bf16(g.z, g.w);
      *(uint2*)(&sB[row * 72 + col]) = e;
      if (kt + 64 < K) __builtin_prefetch(wp + 64, 0, 1);
    }
    __syncthreads();

    // ---- 32 WMMAs per wave per K-tile ----
#pragma unroll
    for (int kk = 0; kk < 64; kk += 32) {
      v16bf a[4], b[4];
#pragma unroll
      for (int i = 0; i < 4; ++i)
        a[i] = load_frag(sA, waveM * 64 + i * 16, 72, kk, lane);
#pragma unroll
      for (int j = 0; j < 4; ++j)
        b[j] = load_frag(sB, waveN * 64 + j * 16, 72, kk, lane);
#pragma unroll
      for (int i = 0; i < 4; ++i)
#pragma unroll
        for (int j = 0; j < 4; ++j)
          acc[i][j] = wmma_bf16(a[i], b[j], acc[i][j]);
    }
  }

  // ---- epilogue: C-fragment element (lane<16 ? r : 8+r, lane&15) ----
  const int rbase = (lane >> 4) << 3;
  const int cloc  = lane & 15;
#pragma unroll
  for (int i = 0; i < 4; ++i)
#pragma unroll
    for (int j = 0; j < 4; ++j)
#pragma unroll
      for (int r = 0; r < 8; ++r) {
        int row = m0 + waveM * 64 + i * 16 + rbase + r;
        int col = n0 + waveN * 64 + j * 16 + cloc;
        if constexpr (OUT_IS_F32)
          ((float*)Cptr)[(size_t)row * N + col] = acc[i][j][r];
        else
          ((unsigned short*)Cptr)[(size_t)row * N + col] = bfu(acc[i][j][r]);
      }
}

// ---------------------------------------------------------------------------
// Flash attention: one block = 64 query rows of one head of one batch.
// 128 threads (4 waves), each wave owns 16 query rows.
// Q,K,V are bf16 from the projection GEMMs. Causal mask applied analytically.
// ---------------------------------------------------------------------------
__global__ __launch_bounds__(128) void attn_kernel(
    const unsigned short* __restrict__ Q,   // [BS, DMODEL]
    const unsigned short* __restrict__ Kb,  // [BS, KVDIM]
    const unsigned short* __restrict__ Vb,  // [BS, KVDIM]
    unsigned short* __restrict__ O)         // [BS, DMODEL]
{
  __shared__ unsigned short sK [64 * 72];     // key-major  [key][feat]
  __shared__ unsigned short sVt[64 * 72];     // transposed [feat][key]
  __shared__ unsigned short sP [4 * 16 * 72]; // per-wave P relayout buffer

  const int t = threadIdx.x, lane = t & 31, w = t >> 5;
  const int qt  = blockIdx.x;               // query tile (64 rows)
  const int h   = blockIdx.y;               // head
  const int b   = blockIdx.z;               // batch
  const int kvh = h / (N_HEADS / (KVDIM / HEAD_DIM)); // h/4 (GQA group)
  const int rowbase = b * SEQ;
  const int q0 = qt * 64;

  const int rl0 = (lane >> 4) << 3;         // C-frag row base for this lane
  const int cl  = lane & 15;                // C-frag col for this lane

  // Q fragments straight from global (row-contiguous)
  const unsigned short* Qp =
      Q + (size_t)(rowbase + q0 + w * 16) * DMODEL + h * HEAD_DIM;
  v16bf qa0 = load_frag(Qp, 0, DMODEL, 0,  lane);
  v16bf qa1 = load_frag(Qp, 0, DMODEL, 32, lane);

  const v8f vzero = {};
  v8f o[4];
#pragma unroll
  for (int nf = 0; nf < 4; ++nf) o[nf] = vzero;
  float mrow[8], lrow[8];
#pragma unroll
  for (int r = 0; r < 8; ++r) { mrow[r] = -1e30f; lrow[r] = 0.f; }

  unsigned short* myP = sP + w * (16 * 72);

  for (int j = 0; j <= qt; ++j) {           // causal: only lower key tiles
    const int kv0 = j * 64;
    __syncthreads();
    // ---- stage K (row-major) and V (transposed) tiles ----
#pragma unroll
    for (int i = 0; i < 8; ++i) {
      int idx = i * 512 + t * 4;            // 64x64 elems
      int row = idx >> 6, col = idx & 63;
      size_t gro = (size_t)(rowbase + kv0 + row) * KVDIM + kvh * HEAD_DIM + col;
      *(uint2*)(&sK[row * 72 + col]) = *(const uint2*)(Kb + gro);
      uint2 v = *(const uint2*)(Vb + gro);
      sVt[(col + 0) * 72 + row] = (unsigned short)(v.x & 0xffffu);
      sVt[(col + 1) * 72 + row] = (unsigned short)(v.x >> 16);
      sVt[(col + 2) * 72 + row] = (unsigned short)(v.y & 0xffffu);
      sVt[(col + 3) * 72 + row] = (unsigned short)(v.y >> 16);
    }
    __syncthreads();

    // ---- scores S = Q * K^T  (per wave: 16 x 64) ----
    v8f s[4];
#pragma unroll
    for (int n = 0; n < 4; ++n) {
      v16bf b0 = load_frag(sK, n * 16, 72, 0,  lane);
      v16bf b1 = load_frag(sK, n * 16, 72, 32, lane);
      v8f c = vzero;
      c = wmma_bf16(qa0, b0, c);
      c = wmma_bf16(qa1, b1, c);
      s[n] = c;
    }
    // scale + causal mask (only the diagonal tile needs masking)
#pragma unroll
    for (int n = 0; n < 4; ++n)
#pragma unroll
      for (int r = 0; r < 8; ++r) {
        float v = s[n][r] * 0.125f;         // 1/sqrt(64)
        if (j == qt) {
          int col = kv0 + n * 16 + cl;
          int row = q0 + w * 16 + rl0 + r;
          if (col > row) v = -1e30f;
        }
        s[n][r] = v;
      }

    // ---- online softmax: rows 0-7 in lanes 0-15, rows 8-15 in lanes 16-31
    float mnew[8], scale[8], psum[8];
#pragma unroll
    for (int r = 0; r < 8; ++r) {
      float m = fmaxf(fmaxf(s[0][r], s[1][r]), fmaxf(s[2][r], s[3][r]));
      m = fmaxf(m, __shfl_xor(m, 1, 32));
      m = fmaxf(m, __shfl_xor(m, 2, 32));
      m = fmaxf(m, __shfl_xor(m, 4, 32));
      m = fmaxf(m, __shfl_xor(m, 8, 32));
      mnew[r]  = fmaxf(mrow[r], m);
      scale[r] = __expf(mrow[r] - mnew[r]);
      mrow[r]  = mnew[r];
      psum[r]  = 0.f;
    }
    // P = exp(S - m), stash into LDS for C-frag -> A-frag relayout
#pragma unroll
    for (int n = 0; n < 4; ++n)
#pragma unroll
      for (int r = 0; r < 8; ++r) {
        float p = __expf(s[n][r] - mnew[r]);
        psum[r] += p;
        myP[(rl0 + r) * 72 + n * 16 + cl] = bfu(p);
      }
#pragma unroll
    for (int r = 0; r < 8; ++r) {
      float ps = psum[r];
      ps += __shfl_xor(ps, 1, 32);
      ps += __shfl_xor(ps, 2, 32);
      ps += __shfl_xor(ps, 4, 32);
      ps += __shfl_xor(ps, 8, 32);
      lrow[r] = lrow[r] * scale[r] + ps;
    }
    // rescale running output
#pragma unroll
    for (int nf = 0; nf < 4; ++nf)
#pragma unroll
      for (int r = 0; r < 8; ++r) o[nf][r] *= scale[r];

    // ---- O += P * V  (wave-private P; same-wave LDS ops are in order) ----
    v16bf pa0 = load_frag(myP, 0, 72, 0,  lane);
    v16bf pa1 = load_frag(myP, 0, 72, 32, lane);
#pragma unroll
    for (int nf = 0; nf < 4; ++nf) {
      v16bf vb0 = load_frag(sVt, nf * 16, 72, 0,  lane);
      v16bf vb1 = load_frag(sVt, nf * 16, 72, 32, lane);
      o[nf] = wmma_bf16(pa0, vb0, o[nf]);
      o[nf] = wmma_bf16(pa1, vb1, o[nf]);
    }
  }

  // ---- normalize + write attention output (bf16) ----
#pragma unroll
  for (int nf = 0; nf < 4; ++nf)
#pragma unroll
    for (int r = 0; r < 8; ++r) {
      int row = rowbase + q0 + w * 16 + rl0 + r;
      int col = h * HEAD_DIM + nf * 16 + cl;
      O[(size_t)row * DMODEL + col] = bfu(o[nf][r] / lrow[r]);
    }
}

// ---------------------------------------------------------------------------
extern "C" void kernel_launch(void* const* d_in, const int* in_sizes, int n_in,
                              void* d_out, int out_size, void* d_ws, size_t ws_size,
                              hipStream_t stream) {
  (void)in_sizes; (void)n_in; (void)out_size; (void)ws_size;
  const float* x  = (const float*)d_in[0];
  const float* wq = (const float*)d_in[1];
  const float* wk = (const float*)d_in[2];
  const float* wv = (const float*)d_in[3];
  const float* wo = (const float*)d_in[4];
  // d_in[5] (mask) is the standard causal mask -> applied analytically.
  // d_in[6] (start_pos) == 0 for prefill.
  float* out = (float*)d_out;

  unsigned short* Qb = (unsigned short*)d_ws;                 // [BS, DMODEL] bf16
  unsigned short* Kb = Qb + (size_t)BS * DMODEL;              // [BS, KVDIM]  bf16
  unsigned short* Vb = Kb + (size_t)BS * KVDIM;               // [BS, KVDIM]  bf16
  unsigned short* Ab = Vb + (size_t)BS * KVDIM;               // [BS, DMODEL] bf16

  dim3 blk(256);
  gemm_wmma_kernel<true,  false><<<dim3(DMODEL/256, BS/128), blk, 0, stream>>>(x,  wq, Qb, BS, DMODEL, DMODEL);
  gemm_wmma_kernel<true,  false><<<dim3(KVDIM /256, BS/128), blk, 0, stream>>>(x,  wk, Kb, BS, KVDIM,  DMODEL);
  gemm_wmma_kernel<true,  false><<<dim3(KVDIM /256, BS/128), blk, 0, stream>>>(x,  wv, Vb, BS, KVDIM,  DMODEL);
  attn_kernel<<<dim3(SEQ/64, N_HEADS, BATCH), dim3(128), 0, stream>>>(Qb, Kb, Vb, Ab);
  gemm_wmma_kernel<false, true ><<<dim3(DMODEL/256, BS/128), blk, 0, stream>>>(Ab, wo, out, BS, DMODEL, DMODEL);
}
